// ImprovedGAT_15960098472724
// MI455X (gfx1250) — matrix operand order repaired and checked by reading, compiled, not run on hardware
//
#include <hip/hip_runtime.h>

typedef __attribute__((ext_vector_type(16))) _Float16 v16h;
typedef __attribute__((ext_vector_type(8)))  float    v8f;

// ---------------- helpers ----------------
static __device__ __forceinline__ unsigned fenc(float f) {
  unsigned u = __float_as_uint(f);
  return (u & 0x80000000u) ? ~u : (u | 0x80000000u);   // monotonic float->uint
}
static __device__ __forceinline__ float fdec(unsigned e) {
  return (e & 0x80000000u) ? __uint_as_float(e & 0x7fffffffu) : __uint_as_float(~e);
}
static __device__ __forceinline__ void edge_ends(const int* __restrict__ src,
                                                 const int* __restrict__ dst,
                                                 int e, int E, int& s, int& d) {
  if (e < E) { s = src[e]; d = dst[e]; } else { s = e - E; d = s; }  // self-loops appended
}

// ---------------- elementwise kernels ----------------
__global__ void fill_u32_kernel(unsigned* __restrict__ p, unsigned v, long n) {
  long i = (long)blockIdx.x * blockDim.x + threadIdx.x;
  if (i < n) p[i] = v;
}

__global__ void f32_to_f16_kernel(const float* __restrict__ s, _Float16* __restrict__ d, long n) {
  long i = (long)blockIdx.x * blockDim.x + threadIdx.x;
  if (i < n) d[i] = (_Float16)s[i];
}

// ---------------- WMMA GEMM: C[M,N] = A[M,K](f16) * B[K,N](f16), f32 accum ----------------
// BM=128, BN=128, BK=32; 8 waves in a 2(row) x 4(col) grid; each wave owns 64x32
// -> 8 x v_wmma_f32_16x16x32_f16 per BK step per wave.
#define BM 128
#define BN 128
#define BK 32

__global__ __launch_bounds__(256)
void gemm_wmma_kernel(const _Float16* __restrict__ A, const _Float16* __restrict__ B,
                      float* __restrict__ C, int M, int K, int N) {
  __shared__ _Float16 As[BM][BK];   // row-major [m][k]
  __shared__ _Float16 Bs[BN][BK];   // transposed: [n][k]

  const int tid  = threadIdx.x;
  const int lane = tid & 31;
  const int wid  = tid >> 5;
  const int bm   = blockIdx.y * BM;
  const int bn   = blockIdx.x * BN;
  const int m0   = (wid & 1) * 64;   // wave row offset (2 row groups)
  const int n0   = (wid >> 1) * 32;  // wave col offset (4 col groups)

  const bool interiorM = (bm + BM) <= M;
  const bool interiorN = (bn + BN) <= N;

  v8f acc[4][2] = {};

  for (int k0 = 0; k0 < K; k0 += BK) {
    // ---- stage A tile (BM x BK): 512 chunks of 8 halfs (16B), 2 per thread ----
    if (interiorM) {
#pragma unroll
      for (int it = 0; it < 2; ++it) {
        int i   = tid + it * 256;
        int row = i >> 2;            // BK/8 = 4 chunks per row
        int col = (i & 3) * 8;
        *(uint4*)&As[row][col] = *(const uint4*)(A + (size_t)(bm + row) * K + k0 + col);
      }
    } else {
#pragma unroll
      for (int it = 0; it < 2; ++it) {
        int i   = tid + it * 256;
        int row = i >> 2;
        int col = (i & 3) * 8;
        uint4 val = make_uint4(0u, 0u, 0u, 0u);
        if (bm + row < M) val = *(const uint4*)(A + (size_t)(bm + row) * K + k0 + col);
        *(uint4*)&As[row][col] = val;
      }
    }
    // ---- stage B tile (BK x BN) transposed into Bs[n][k]: 512 chunks, 2 per thread ----
#pragma unroll
    for (int it = 0; it < 2; ++it) {
      int i  = tid + it * 256;
      int kr = i >> 4;               // BN/8 = 16 chunks per K row
      int nc = (i & 15) * 8;
      const _Float16* srcp = B + (size_t)(k0 + kr) * N + bn + nc;
      if (interiorN || (bn + nc + 8 <= N)) {
        uint4 val = *(const uint4*)srcp;
        const _Float16* hv = (const _Float16*)&val;
#pragma unroll
        for (int j = 0; j < 8; ++j) Bs[nc + j][kr] = hv[j];
      } else {
#pragma unroll
        for (int j = 0; j < 8; ++j)
          Bs[nc + j][kr] = (bn + nc + j < N) ? srcp[j] : (_Float16)0.0f;
      }
    }
    __syncthreads();

    // ---- fragment loads per ISA layout ----
    // A: lane = 16*(khalf) + m ; halves 0..7 -> K = 8*khalf + j ; halves 8..15 -> K = 16 + 8*khalf + (j-8)
    // B: lane = 16*(khalf) + n ; halves 0..7 -> K = 8*khalf + j ; halves 8..15 -> K = 16 + 8*khalf + (j-8)
    const int lm  = lane & 15;
    const int kb  = (lane >> 4) * 8;
    const int kbb = (lane >> 4) * 8;
    v16h af[4], bf[2];
#pragma unroll
    for (int t = 0; t < 4; ++t) {
      uint4* p = (uint4*)&af[t];
      p[0] = *(const uint4*)&As[m0 + t * 16 + lm][kb];
      p[1] = *(const uint4*)&As[m0 + t * 16 + lm][kb + 16];
    }
#pragma unroll
    for (int t = 0; t < 2; ++t) {
      uint4* q = (uint4*)&bf[t];
      q[0] = *(const uint4*)&Bs[n0 + t * 16 + lm][kbb];
      q[1] = *(const uint4*)&Bs[n0 + t * 16 + lm][kbb + 16];
    }
#pragma unroll
    for (int i = 0; i < 4; ++i)
#pragma unroll
      for (int j = 0; j < 2; ++j)
        acc[i][j] = __builtin_amdgcn_wmma_f32_16x16x32_f16(
            false, af[i], false, bf[j], (short)0, acc[i][j], false, false);
    __syncthreads();
  }

  // ---- store D: VGPR r -> row = base + (lane>=16)*8 + r, col = base + (lane&15) ----
#pragma unroll
  for (int i = 0; i < 4; ++i) {
#pragma unroll
    for (int j = 0; j < 2; ++j) {
      int col = bn + n0 + j * 16 + (lane & 15);
      if (col >= N) continue;
      int rbase = bm + m0 + i * 16 + ((lane >> 4) * 8);
#pragma unroll
      for (int r = 0; r < 8; ++r) {
        int row = rbase + r;
        if (row < M) C[(size_t)row * N + col] = acc[i][j][r];
      }
    }
  }
}

// ---------------- attention kernels ----------------
__global__ void alpha_kernel(const float* __restrict__ h,
                             const float* __restrict__ a_src,
                             const float* __restrict__ a_dst,
                             float* __restrict__ as_, float* __restrict__ ad_,
                             int N, int H, int C) {
  int idx = blockIdx.x * blockDim.x + threadIdx.x;
  if (idx >= N * H) return;
  int n = idx / H, hd = idx % H;
  const float* hp = h + (size_t)n * H * C + (size_t)hd * C;
  const float* sp = a_src + hd * C;
  const float* dp = a_dst + hd * C;
  float s1 = 0.f, s2 = 0.f;
  for (int c = 0; c < C; ++c) { float v = hp[c]; s1 += v * sp[c]; s2 += v * dp[c]; }
  as_[idx] = s1; ad_[idx] = s2;
}

__global__ void edge_max_kernel(const int* __restrict__ src, const int* __restrict__ dst,
                                const float* __restrict__ as_, const float* __restrict__ ad_,
                                unsigned* __restrict__ menc, int E, int Etot, int H) {
  long idx = (long)blockIdx.x * blockDim.x + threadIdx.x;
  if (idx >= (long)Etot * H) return;
  int e = (int)(idx / H), hd = (int)(idx % H);
  int s, d; edge_ends(src, dst, e, E, s, d);
  float v = as_[s * H + hd] + ad_[d * H + hd];
  v = v > 0.f ? v : 0.2f * v;                       // leaky_relu(0.2)
  atomicMax(&menc[d * H + hd], fenc(v));
}

__global__ void edge_exp_kernel(const int* __restrict__ src, const int* __restrict__ dst,
                                const float* __restrict__ as_, const float* __restrict__ ad_,
                                const unsigned* __restrict__ menc, float* __restrict__ ssum,
                                float* __restrict__ exb, int E, int Etot, int H) {
  long idx = (long)blockIdx.x * blockDim.x + threadIdx.x;
  if (idx >= (long)Etot * H) return;
  int e = (int)(idx / H), hd = (int)(idx % H);
  int s, d; edge_ends(src, dst, e, E, s, d);
  float v = as_[s * H + hd] + ad_[d * H + hd];
  v = v > 0.f ? v : 0.2f * v;
  float ex = expf(v - fdec(menc[d * H + hd]));
  exb[idx] = ex;
  atomicAdd(&ssum[d * H + hd], ex);
}

__global__ void edge_norm_kernel(const int* __restrict__ src, const int* __restrict__ dst,
                                 const float* __restrict__ ssum, float* __restrict__ exb,
                                 int E, int Etot, int H) {
  long idx = (long)blockIdx.x * blockDim.x + threadIdx.x;
  if (idx >= (long)Etot * H) return;
  int e = (int)(idx / H), hd = (int)(idx % H);
  int s, d; edge_ends(src, dst, e, E, s, d);
  exb[idx] = exb[idx] / (ssum[d * H + hd] + 1e-16f);
}

__global__ void edge_scatter_kernel(const int* __restrict__ src, const int* __restrict__ dst,
                                    const float* __restrict__ exb, const float* __restrict__ h,
                                    float* __restrict__ out, int E, int Etot, int H, int C) {
  long idx = (long)blockIdx.x * blockDim.x + threadIdx.x;
  int OC = H * C;
  if (idx >= (long)Etot * OC) return;
  int e   = (int)(idx / OC);
  int rem = (int)(idx % OC);
  int hd  = rem / C;
  int s, d; edge_ends(src, dst, e, E, s, d);
  float a = exb[(long)e * H + hd];
  atomicAdd(&out[(size_t)d * OC + rem], a * h[(size_t)s * OC + rem]);
}

__global__ void bias_relu_kernel(float* __restrict__ o, const float* __restrict__ b,
                                 long n, int OC) {
  long idx = (long)blockIdx.x * blockDim.x + threadIdx.x;
  if (idx >= n) return;
  o[idx] = fmaxf(o[idx] + b[idx % OC], 0.f);
}

__global__ void bias_copy_kernel(const float* __restrict__ o, const float* __restrict__ b,
                                 float* __restrict__ z, long n, int OC) {
  long idx = (long)blockIdx.x * blockDim.x + threadIdx.x;
  if (idx >= n) return;
  z[idx] = o[idx] + b[idx % OC];     // heads==1: mean over heads is identity, no relu
}

__global__ void skip_kernel(const float* __restrict__ x, const float* __restrict__ Wsk,
                            const float* __restrict__ bsk, float* __restrict__ z, int N) {
  int idx = blockIdx.x * blockDim.x + threadIdx.x;
  if (idx >= N * 16) return;
  int n = idx >> 4, c = idx & 15;
  float acc = z[idx] + bsk[c];
  const float* xp = x + (size_t)n * 128;
  for (int k = 0; k < 128; ++k) acc += xp[k] * Wsk[k * 16 + c];
  z[idx] = acc;
}

__global__ void logsoftmax_kernel(const float* __restrict__ z, float* __restrict__ out, int N) {
  int n = blockIdx.x * blockDim.x + threadIdx.x;
  if (n >= N) return;
  const float* zp = z + (size_t)n * 16;
  float m = zp[0];
  for (int c = 1; c < 16; ++c) m = fmaxf(m, zp[c]);
  float s = 0.f;
  for (int c = 0; c < 16; ++c) s += expf(zp[c] - m);
  float l = logf(s);
  for (int c = 0; c < 16; ++c) out[n * 16 + c] = zp[c] - m - l;
}

// ---------------- launcher ----------------
static inline unsigned gblk(long n) { return (unsigned)((n + 255) / 256); }

extern "C" void kernel_launch(void* const* d_in, const int* in_sizes, int n_in,
                              void* d_out, int out_size, void* d_ws, size_t ws_size,
                              hipStream_t stream) {
  (void)n_in; (void)out_size; (void)ws_size;

  const float* x    = (const float*)d_in[0];
  const int*   srcI = (const int*)d_in[1];
  const int*   dstI = (const int*)d_in[2];
  const float* W1 = (const float*)d_in[3],  *as1 = (const float*)d_in[4],
             * ad1 = (const float*)d_in[5], *b1  = (const float*)d_in[6];
  const float* W2 = (const float*)d_in[7],  *as2 = (const float*)d_in[8],
             * ad2 = (const float*)d_in[9], *b2  = (const float*)d_in[10];
  const float* W3 = (const float*)d_in[11], *as3 = (const float*)d_in[12],
             * ad3 = (const float*)d_in[13],*b3  = (const float*)d_in[14];
  const float* W4 = (const float*)d_in[15], *as4 = (const float*)d_in[16],
             * ad4 = (const float*)d_in[17],*b4  = (const float*)d_in[18];
  const float* Wsk = (const float*)d_in[19],*bsk = (const float*)d_in[20];

  const int N    = in_sizes[0] / 128;
  const int E    = in_sizes[1];
  const int Etot = E + N;

  // carve workspace
  char*  ws  = (char*)d_ws;
  size_t off = 0;
  auto carve = [&](size_t bytes) -> void* {
    void* p = ws + off;
    off = (off + bytes + 255) & ~(size_t)255;
    return p;
  };
  float*     hbuf = (float*)    carve((size_t)N * 2048 * 4);
  float*     obuf = (float*)    carve((size_t)N * 2048 * 4);
  _Float16*  a16  = (_Float16*) carve((size_t)N * 2048 * 2);
  _Float16*  w16  = (_Float16*) carve((size_t)2048 * 1024 * 2);
  float*     asb  = (float*)    carve((size_t)N * 16 * 4);
  float*     adb  = (float*)    carve((size_t)N * 16 * 4);
  unsigned*  menc = (unsigned*) carve((size_t)N * 16 * 4);
  float*     ssum = (float*)    carve((size_t)N * 16 * 4);
  float*     exb  = (float*)    carve((size_t)Etot * 16 * 4);
  float*     x4b  = (float*)    carve((size_t)N * 16 * 4);

  struct L { const float *W, *as, *ad, *b; int IC, H, C; };
  L layers[4] = {
    { W1, as1, ad1, b1,  128, 16, 128 },
    { W2, as2, ad2, b2, 2048, 16,  64 },
    { W3, as3, ad3, b3, 1024, 16,  32 },
    { W4, as4, ad4, b4,  512,  1,  16 },
  };

  const float* act = x;
  for (int li = 0; li < 4; ++li) {
    const L& P = layers[li];
    const int OC = P.H * P.C;

    // 1) convert activation + weight to f16 (activation consumed before obuf reuse)
    long nIn = (long)N * P.IC;
    f32_to_f16_kernel<<<gblk(nIn), 256, 0, stream>>>(act, a16, nIn);
    long nW = (long)P.IC * OC;
    f32_to_f16_kernel<<<gblk(nW), 256, 0, stream>>>(P.W, w16, nW);

    // 2) h = act @ W via WMMA
    dim3 g((OC + BN - 1) / BN, (N + BM - 1) / BM);
    gemm_wmma_kernel<<<g, 256, 0, stream>>>(a16, w16, hbuf, N, P.IC, OC);

    // 3) per-(node,head) attention logits
    long NH = (long)N * P.H;
    alpha_kernel<<<gblk(NH), 256, 0, stream>>>(hbuf, P.as, P.ad, asb, adb, N, P.H, P.C);

    // 4) init stats + zero aggregation target (act already consumed into a16)
    fill_u32_kernel<<<gblk(NH), 256, 0, stream>>>(menc, 0u, NH);
    fill_u32_kernel<<<gblk(NH), 256, 0, stream>>>((unsigned*)ssum, 0u, NH);
    fill_u32_kernel<<<gblk((long)N * OC), 256, 0, stream>>>((unsigned*)obuf, 0u, (long)N * OC);

    // 5) segment softmax over edges (incl. self-loops)
    long EH = (long)Etot * P.H;
    edge_max_kernel<<<gblk(EH), 256, 0, stream>>>(srcI, dstI, asb, adb, menc, E, Etot, P.H);
    edge_exp_kernel<<<gblk(EH), 256, 0, stream>>>(srcI, dstI, asb, adb, menc, ssum, exb, E, Etot, P.H);
    edge_norm_kernel<<<gblk(EH), 256, 0, stream>>>(srcI, dstI, ssum, exb, E, Etot, P.H);

    // 6) weighted message scatter
    long EHC = (long)Etot * OC;
    edge_scatter_kernel<<<gblk(EHC), 256, 0, stream>>>(srcI, dstI, exb, hbuf, obuf, E, Etot, P.H, P.C);

    // 7) bias (+relu for concat layers)
    if (li < 3) {
      bias_relu_kernel<<<gblk((long)N * OC), 256, 0, stream>>>(obuf, P.b, (long)N * OC, OC);
      act = obuf;
    } else {
      bias_copy_kernel<<<gblk((long)N * OC), 256, 0, stream>>>(obuf, P.b, x4b, (long)N * OC, OC);
    }
  }

  // skip connection + log-softmax
  skip_kernel<<<gblk((long)N * 16), 256, 0, stream>>>(x, Wsk, bsk, x4b, N);
  logsoftmax_kernel<<<gblk(N), 256, 0, stream>>>(x4b, (float*)d_out, N);
}
